// InnerSelfAttention_19576460935753
// MI455X (gfx1250) — compile-verified, matrix operand order checked by reading, not run on hardware
//
#include <hip/hip_runtime.h>
#include <hip/hip_bf16.h>

// ---------------------------------------------------------------------------
// Shapes (fixed by the reference)
// ---------------------------------------------------------------------------
#define BB 2
#define SS 2048
#define DD 1024
#define HH 16
#define HDIM 64
#define PP 512
#define MTOT (BB * SS)        // 4096 rows
#define BIGNEG -3.0e38f

typedef __attribute__((ext_vector_type(16))) __bf16 v16bf;
typedef __attribute__((ext_vector_type(8)))  float  v8f;
typedef __attribute__((ext_vector_type(4)))  unsigned int v4u;
typedef __attribute__((ext_vector_type(8)))  int v8i;
typedef __attribute__((ext_vector_type(4)))  int v4i;

__device__ __forceinline__ unsigned short f2bf(float f) {
  unsigned int u = __float_as_uint(f);
  unsigned int r = u + 0x7FFFu + ((u >> 16) & 1u);   // round-to-nearest-even
  return (unsigned short)(r >> 16);
}

__device__ __forceinline__ v8f wmma_bf16(v16bf a, v16bf b, v8f c) {
  return __builtin_amdgcn_wmma_f32_16x16x32_bf16(false, a, false, b,
                                                 (short)0, c, false, false);
}

// ---------------------------------------------------------------------------
// Tensor Data Mover: DMA a 2D bf16 tile [tile_y rows x tile_x elems] from
// global (row stride = stride_x elems) into LDS, packed row-major.
// D# bit layout per CDNA5 ISA ch.8 (groups 0/1; groups 2/3 zero for 2D).
// ---------------------------------------------------------------------------
__device__ __forceinline__ void tdm_load_2d(void* lds_ptr, const void* gptr,
                                            unsigned tile_x, unsigned tile_y,
                                            unsigned long long stride_x) {
  unsigned long long ga = (unsigned long long)(uintptr_t)gptr;
  v4u g0;
  g0[0] = 1u;                                      // count=1, user descriptor
  g0[1] = (unsigned)(uintptr_t)lds_ptr;            // lds_addr (bytes)
  g0[2] = (unsigned)(ga & 0xFFFFFFFFu);            // global_addr[31:0]
  g0[3] = (unsigned)((ga >> 32) & 0x01FFFFFFu)     // global_addr[56:32]
        | (2u << 30);                              // type = 2 ("image")
  // group1: data_size=1 (2 bytes); tensor dims == tile dims; dim0 stride
  unsigned long long q0 = (1ull << 16)
                        | ((unsigned long long)(tile_x & 0xFFFFu) << 48);
  unsigned long long q1 = (unsigned long long)(tile_x >> 16)
                        | ((unsigned long long)tile_y << 16)        // tensor_dim1
                        | ((unsigned long long)(tile_x & 0xFFFFu) << 48); // tile_dim0
  unsigned long long q2 = (unsigned long long)(tile_y & 0xFFFFu)    // tile_dim1
                        | ((stride_x & 0xFFFFFFFFull) << 32);       // stride[31:0]
  unsigned long long q3 = (stride_x >> 32) & 0xFFFFull;             // stride[47:32]
  v8i g1;
  g1[0] = (int)(unsigned)q0;  g1[1] = (int)(unsigned)(q0 >> 32);
  g1[2] = (int)(unsigned)q1;  g1[3] = (int)(unsigned)(q1 >> 32);
  g1[4] = (int)(unsigned)q2;  g1[5] = (int)(unsigned)(q2 >> 32);
  g1[6] = (int)(unsigned)q3;  g1[7] = (int)(unsigned)(q3 >> 32);
  v4i z4 = {0, 0, 0, 0};
#if defined(__clang_major__) && __clang_major__ >= 23
  v8i z8 = {0, 0, 0, 0, 0, 0, 0, 0};
  __builtin_amdgcn_tensor_load_to_lds(g0, g1, z4, z4, z8, 0);
#else
  __builtin_amdgcn_tensor_load_to_lds(g0, g1, z4, z4, 0);
#endif
}

// ---------------------------------------------------------------------------
// Kernel 1: f32 -> bf16 (hidden states), 4 elems / thread
// ---------------------------------------------------------------------------
__global__ __launch_bounds__(256) void cvt_x(const float* __restrict__ x,
                                             unsigned short* __restrict__ o) {
  int i = (blockIdx.x * 256 + threadIdx.x) * 4;
  float4 v = *(const float4*)(x + i);
  unsigned short r[4] = {f2bf(v.x), f2bf(v.y), f2bf(v.z), f2bf(v.w)};
  *(uint2*)(o + i) = *(const uint2*)r;
}

// ---------------------------------------------------------------------------
// Kernel 2: weights f32 [K][N] -> bf16 transposed [N][K]
// ---------------------------------------------------------------------------
__global__ __launch_bounds__(256) void cvt_wt(
    const float* __restrict__ W0, const float* __restrict__ W1,
    const float* __restrict__ W2, const float* __restrict__ W3,
    unsigned short* __restrict__ T0, unsigned short* __restrict__ T1,
    unsigned short* __restrict__ T2, unsigned short* __restrict__ T3) {
  const float* W = (blockIdx.z == 0) ? W0 : (blockIdx.z == 1) ? W1
                 : (blockIdx.z == 2) ? W2 : W3;
  unsigned short* T = (blockIdx.z == 0) ? T0 : (blockIdx.z == 1) ? T1
                    : (blockIdx.z == 2) ? T2 : T3;
  int n = blockIdx.x * 16 + threadIdx.x;
  int k = blockIdx.y * 16 + threadIdx.y;
  T[n * DD + k] = f2bf(W[k * DD + n]);
}

// ---------------------------------------------------------------------------
// Kernel 3: QKV GEMM.  C[4096,1024] = Xbf[4096,1024] * W (W given transposed).
// WG tile 128(M) x 64(N); 8 waves, each wave 16(M) x 64(N); K-step 32.
// TDM double-buffered LDS staging: wave 0 DMAs tile k+1 while all compute k.
// grid.z selects Wq/Wk/Wv and the destination.
// ---------------------------------------------------------------------------
__global__ __launch_bounds__(256) void gemm_qkv(
    const unsigned short* __restrict__ X,
    const unsigned short* __restrict__ Wqt,
    const unsigned short* __restrict__ Wkt,
    const unsigned short* __restrict__ Wvt,
    unsigned short* __restrict__ Qo,
    unsigned short* __restrict__ Ko,
    unsigned short* __restrict__ Vo) {
  const unsigned short* W = (blockIdx.z == 0) ? Wqt : (blockIdx.z == 1) ? Wkt : Wvt;
  unsigned short* Out     = (blockIdx.z == 0) ? Qo  : (blockIdx.z == 1) ? Ko  : Vo;

  __shared__ __align__(32) unsigned short sA[2][128 * 32];
  __shared__ __align__(32) unsigned short sB[2][64 * 32];

  int n0 = blockIdx.x * 64;
  int m0 = blockIdx.y * 128;
  int w = threadIdx.x >> 5, lane = threadIdx.x & 31;
  int half = lane >> 4, ln = lane & 15;

  v8f acc[4] = {};

  // prologue: DMA tile 0
  if (w == 0) {
    tdm_load_2d(&sA[0][0], X + (size_t)m0 * DD, 32, 128, DD);
    tdm_load_2d(&sB[0][0], W + (size_t)n0 * DD, 32, 64, DD);
  }
  int cur = 0;
  for (int k0 = 0; k0 < DD; k0 += 32) {
    if (w == 0) __builtin_amdgcn_s_wait_tensorcnt(0);  // tile k0 resident
    __syncthreads();
    if (w == 0 && (k0 + 32) < DD) {                    // DMA tile k0+32
      tdm_load_2d(&sA[cur ^ 1][0], X + (size_t)m0 * DD + k0 + 32, 32, 128, DD);
      tdm_load_2d(&sB[cur ^ 1][0], W + (size_t)n0 * DD + k0 + 32, 32, 64, DD);
    }
    // ---- compute: 1 A-frag, 4 B-frags, 4 WMMAs ----
    v16bf a = *(const v16bf*)(&sA[cur][(w * 16 + ln) * 32 + half * 16]);
#pragma unroll
    for (int nt = 0; nt < 4; ++nt) {
      v16bf b = *(const v16bf*)(&sB[cur][(nt * 16 + ln) * 32 + half * 16]);
      acc[nt] = wmma_bf16(a, b, acc[nt]);
    }
    cur ^= 1;
  }

  // ---- epilogue: bf16 store [4096][1024] == [b,s,h,hd] ----
#pragma unroll
  for (int nt = 0; nt < 4; ++nt)
#pragma unroll
    for (int r = 0; r < 8; ++r) {
      int m = m0 + w * 16 + r + 8 * half;
      int n = n0 + nt * 16 + ln;
      Out[(size_t)m * DD + n] = f2bf(acc[nt][r]);
    }
}

// ---------------------------------------------------------------------------
// Kernel 4: flash attention per (b,h, 128-row i-block).
// 8 waves x 16-row i-tiles; causal j-sweep in blocks of 32.  K tile staged by
// TDM, V^T staged cooperatively; online softmax; relative-position bias.
// ---------------------------------------------------------------------------
__global__ __launch_bounds__(256) void attn(
    const unsigned short* __restrict__ Q,
    const unsigned short* __restrict__ K,
    const unsigned short* __restrict__ V,
    const float* __restrict__ pos_emb,
    unsigned short* __restrict__ Oa) {
  int bh = blockIdx.x;                // 0..31
  int b = bh >> 4, h = bh & 15;
  int iblk = blockIdx.y;              // 0..15
  int i0wg = iblk * 128;
  int w = threadIdx.x >> 5, lane = threadIdx.x & 31;
  int half = lane >> 4, ln = lane & 15;
  int i0 = i0wg + w * 16;             // wave's i-tile base

  __shared__ __align__(32) float          sBias[2 * PP + 8];
  __shared__ __align__(32) unsigned short sK[32 * 64];     // [j][d]
  __shared__ __align__(32) unsigned short sV[64 * 32];     // [d][j]  (V^T)
  __shared__ __align__(32) unsigned short sP[8][16 * 32];  // per-wave probs

  for (int r = threadIdx.x; r < 2 * PP + 1; r += 256)
    sBias[r] = pos_emb[r * HH + h];

  // Q A-fragments: row i0+ln, d-halves; held across the whole j sweep
  const unsigned short* Qrow = Q + (size_t)(b * SS + i0 + ln) * DD + h * HDIM;
  v16bf aQ0 = *(const v16bf*)(Qrow + half * 16);
  v16bf aQ1 = *(const v16bf*)(Qrow + 32 + half * 16);

  v8f O0 = {}, O1 = {}, O2 = {}, O3 = {};
  float mrun[8], lrun[8];
#pragma unroll
  for (int r = 0; r < 8; ++r) { mrun[r] = BIGNEG; lrun[r] = 0.f; }

  int njb = iblk * 4 + 4;             // j-blocks of 32 up through the diagonal
  for (int jb = 0; jb < njb; ++jb) {
    int j0 = jb * 32;
    __syncthreads();                  // prior-iter LDS readers done
    // ---- stage K tile [32][64] via TDM; V^T tile [64][32] cooperatively ----
    if (w == 0)
      tdm_load_2d(&sK[0], K + (size_t)(b * SS + j0) * DD + h * HDIM, 64, 32, DD);
    {
      int t = threadIdx.x;
      int jj = t >> 3, dd = (t & 7) * 8;
      size_t grow = (size_t)(b * SS + j0 + jj) * DD + h * HDIM + dd;
      uint4 vv = *(const uint4*)(V + grow);
      const unsigned short* ve = (const unsigned short*)&vv;
#pragma unroll
      for (int e = 0; e < 8; ++e) sV[(dd + e) * 32 + jj] = ve[e];
      if (jb + 1 < njb)
        __builtin_prefetch(V + grow + (size_t)32 * DD, 0, 0);
    }
    if (w == 0) __builtin_amdgcn_s_wait_tensorcnt(0);
    __syncthreads();
    if (j0 > i0 + 15) continue;       // fully masked for this wave

    // ---- scores: two 16x16 tiles, K-dim = 64 (2 WMMAs each) ----
    v8f s[2];
#pragma unroll
    for (int nt = 0; nt < 2; ++nt) {
      v8f a2 = {};
      v16bf bk0 = *(const v16bf*)(&sK[(nt * 16 + ln) * 64 + half * 16]);
      a2 = wmma_bf16(aQ0, bk0, a2);
      v16bf bk1 = *(const v16bf*)(&sK[(nt * 16 + ln) * 64 + 32 + half * 16]);
      a2 = wmma_bf16(aQ1, bk1, a2);
      s[nt] = a2;
    }

    // ---- scale + rel-pos bias + causal mask, track tile row-max ----
    float tmax[8];
#pragma unroll
    for (int r = 0; r < 8; ++r) tmax[r] = BIGNEG;
#pragma unroll
    for (int nt = 0; nt < 2; ++nt)
#pragma unroll
      for (int r = 0; r < 8; ++r) {
        int i = i0 + r + 8 * half;
        int j = j0 + nt * 16 + ln;
        float v;
        if (j > i) {
          v = BIGNEG;
        } else {
          int rel = j - i;            // <= 0
          if (rel < -PP) rel = -PP;
          v = s[nt][r] * 0.125f + 1e-9f + sBias[rel + PP];
        }
        s[nt][r] = v;
        tmax[r] = fmaxf(tmax[r], v);
      }
    // row-max across the 16 lanes of this half-wave
#pragma unroll
    for (int msk = 1; msk < 16; msk <<= 1)
#pragma unroll
      for (int r = 0; r < 8; ++r)
        tmax[r] = fmaxf(tmax[r], __shfl_xor(tmax[r], msk, 32));

    // ---- online softmax update ----
    float scale[8];
#pragma unroll
    for (int r = 0; r < 8; ++r) {
      float mn = fmaxf(mrun[r], tmax[r]);
      scale[r] = __expf(mrun[r] - mn);
      mrun[r] = mn;
    }
#pragma unroll
    for (int r = 0; r < 8; ++r) {
      O0[r] *= scale[r]; O1[r] *= scale[r];
      O2[r] *= scale[r]; O3[r] *= scale[r];
    }
    float rsum[8];
#pragma unroll
    for (int r = 0; r < 8; ++r) rsum[r] = 0.f;
#pragma unroll
    for (int nt = 0; nt < 2; ++nt)
#pragma unroll
      for (int r = 0; r < 8; ++r) {
        float p = __expf(s[nt][r] - mrun[r]);
        rsum[r] += p;
        sP[w][(r + 8 * half) * 32 + nt * 16 + ln] = f2bf(p);
      }
#pragma unroll
    for (int msk = 1; msk < 16; msk <<= 1)
#pragma unroll
      for (int r = 0; r < 8; ++r)
        rsum[r] += __shfl_xor(rsum[r], msk, 32);
#pragma unroll
    for (int r = 0; r < 8; ++r) lrun[r] = lrun[r] * scale[r] + rsum[r];

    // ---- P x V: reload probs in A-layout, 4 d-tiles ----
    v16bf ap = *(const v16bf*)(&sP[w][ln * 32 + half * 16]);
    v16bf bv0 = *(const v16bf*)(&sV[(0 * 16 + ln) * 32 + half * 16]);
    O0 = wmma_bf16(ap, bv0, O0);
    v16bf bv1 = *(const v16bf*)(&sV[(1 * 16 + ln) * 32 + half * 16]);
    O1 = wmma_bf16(ap, bv1, O1);
    v16bf bv2 = *(const v16bf*)(&sV[(2 * 16 + ln) * 32 + half * 16]);
    O2 = wmma_bf16(ap, bv2, O2);
    v16bf bv3 = *(const v16bf*)(&sV[(3 * 16 + ln) * 32 + half * 16]);
    O3 = wmma_bf16(ap, bv3, O3);
  }

  // ---- normalize and store merged-head bf16 output [b,s,h*64+d] ----
#pragma unroll
  for (int r = 0; r < 8; ++r) {
    float inv = 1.0f / lrun[r];
    size_t row = (size_t)(b * SS + i0 + r + 8 * half);
    unsigned short* dst = Oa + row * DD + h * HDIM + ln;
    dst[0]  = f2bf(O0[r] * inv);
    dst[16] = f2bf(O1[r] * inv);
    dst[32] = f2bf(O2[r] * inv);
    dst[48] = f2bf(O3[r] * inv);
  }
}

// ---------------------------------------------------------------------------
// Kernel 5: output projection.  out[4096,1024] = attn * Wo^T + bo  (f32 out)
// Same TDM double-buffered structure as gemm_qkv.
// ---------------------------------------------------------------------------
__global__ __launch_bounds__(256) void gemm_out(
    const unsigned short* __restrict__ A,
    const unsigned short* __restrict__ Wot,
    const float* __restrict__ bo,
    float* __restrict__ Out) {
  __shared__ __align__(32) unsigned short sA[2][128 * 32];
  __shared__ __align__(32) unsigned short sB[2][64 * 32];

  int n0 = blockIdx.x * 64;
  int m0 = blockIdx.y * 128;
  int w = threadIdx.x >> 5, lane = threadIdx.x & 31;
  int half = lane >> 4, ln = lane & 15;

  v8f acc[4] = {};

  if (w == 0) {
    tdm_load_2d(&sA[0][0], A + (size_t)m0 * DD, 32, 128, DD);
    tdm_load_2d(&sB[0][0], Wot + (size_t)n0 * DD, 32, 64, DD);
  }
  int cur = 0;
  for (int k0 = 0; k0 < DD; k0 += 32) {
    if (w == 0) __builtin_amdgcn_s_wait_tensorcnt(0);
    __syncthreads();
    if (w == 0 && (k0 + 32) < DD) {
      tdm_load_2d(&sA[cur ^ 1][0], A + (size_t)m0 * DD + k0 + 32, 32, 128, DD);
      tdm_load_2d(&sB[cur ^ 1][0], Wot + (size_t)n0 * DD + k0 + 32, 32, 64, DD);
    }
    v16bf a = *(const v16bf*)(&sA[cur][(w * 16 + ln) * 32 + half * 16]);
#pragma unroll
    for (int nt = 0; nt < 4; ++nt) {
      v16bf b = *(const v16bf*)(&sB[cur][(nt * 16 + ln) * 32 + half * 16]);
      acc[nt] = wmma_bf16(a, b, acc[nt]);
    }
    cur ^= 1;
  }

#pragma unroll
  for (int nt = 0; nt < 4; ++nt) {
    int n = n0 + nt * 16 + ln;
    float bias = bo[n];
#pragma unroll
    for (int r = 0; r < 8; ++r) {
      int m = m0 + w * 16 + r + 8 * half;
      Out[(size_t)m * DD + n] = acc[nt][r] + bias;
    }
  }
}

// ---------------------------------------------------------------------------
// Host-side launcher
// ---------------------------------------------------------------------------
extern "C" void kernel_launch(void* const* d_in, const int* in_sizes, int n_in,
                              void* d_out, int out_size, void* d_ws, size_t ws_size,
                              hipStream_t stream) {
  const float* x       = (const float*)d_in[0];
  const float* Wq      = (const float*)d_in[1];
  const float* Wk      = (const float*)d_in[2];
  const float* Wv      = (const float*)d_in[3];
  const float* Wo      = (const float*)d_in[4];
  const float* bo      = (const float*)d_in[5];
  const float* pos_emb = (const float*)d_in[6];

  char* ws = (char*)d_ws;
  const size_t MiB = 1u << 20;
  unsigned short* Xbf = (unsigned short*)(ws);             //  8 MiB
  unsigned short* Wqt = (unsigned short*)(ws +  8 * MiB);  //  2 MiB
  unsigned short* Wkt = (unsigned short*)(ws + 10 * MiB);  //  2 MiB
  unsigned short* Wvt = (unsigned short*)(ws + 12 * MiB);  //  2 MiB
  unsigned short* Wot = (unsigned short*)(ws + 14 * MiB);  //  2 MiB
  unsigned short* Qb  = (unsigned short*)(ws + 16 * MiB);  //  8 MiB
  unsigned short* Kb  = (unsigned short*)(ws + 24 * MiB);  //  8 MiB
  unsigned short* Vb  = (unsigned short*)(ws + 32 * MiB);  //  8 MiB
  unsigned short* At  = (unsigned short*)(ws + 40 * MiB);  //  8 MiB

  // 1) precision conversion
  cvt_x<<<(MTOT * DD) / (256 * 4), 256, 0, stream>>>(x, Xbf);
  cvt_wt<<<dim3(DD / 16, DD / 16, 4), dim3(16, 16), 0, stream>>>(
      Wq, Wk, Wv, Wo, Wqt, Wkt, Wvt, Wot);

  // 2) fused QKV projection (grid.z picks the weight)
  gemm_qkv<<<dim3(DD / 64, MTOT / 128, 3), 256, 0, stream>>>(
      Xbf, Wqt, Wkt, Wvt, Qb, Kb, Vb);

  // 3) causal flash attention with relative-position bias
  attn<<<dim3(BB * HH, SS / 128), 256, 0, stream>>>(Qb, Kb, Vb, pos_emb, At);

  // 4) output projection + bias (f32 result)
  gemm_out<<<dim3(DD / 64, MTOT / 128), 256, 0, stream>>>(
      At, Wot, bo, (float*)d_out);
}
